// Head_8504035246117
// MI455X (gfx1250) — compile-verified, matrix operand order checked by reading, not run on hardware
//
#include <hip/hip_runtime.h>
#include <hip/hip_bf16.h>

#define Bn 4
#define Tn 4096
#define Cn 1024
#define Hn 64

typedef _Float16 v16h __attribute__((ext_vector_type(16)));
typedef float    v8f  __attribute__((ext_vector_type(8)));
typedef _Float16 h2   __attribute__((ext_vector_type(2)));
typedef float    f2   __attribute__((ext_vector_type(2)));

// K-striping within a 16-bit 16x32 A / 32x16 B fragment (ISA 7.12.2):
// halves e=2*vg+bit; vg<4 -> K = hl*8 + vg*2 + bit ; vg>=4 -> K = 16 + hl*8 + (vg-4)*2 + bit
__device__ __forceinline__ int kbase(int hl, int vg) {
    return (vg < 4) ? (hl * 8 + vg * 2) : (16 + hl * 8 + (vg - 4) * 2);
}

__device__ __forceinline__ v8f wmma_f16(v16h a, v16h b, v8f c) {
    return __builtin_amdgcn_wmma_f32_16x16x32_f16(false, a, false, b, (short)0, c, false, false);
}

// ---------------------------------------------------------------------------
// Kernel 0: repack Wk/Wq/Wv (C x H, f32 row-major) into per-lane B-fragment
// layout: [mat(3)][ntile(4)][kstep(32)][lane(32)][16 halves]
// ---------------------------------------------------------------------------
__global__ void repack_w(const float* __restrict__ Wk, const float* __restrict__ Wq,
                         const float* __restrict__ Wv, _Float16* __restrict__ wpack) {
    int tid = blockIdx.x * blockDim.x + threadIdx.x;   // 3*4*32*32 = 12288 total
    if (tid >= 3 * 4 * 32 * 32) return;
    int lane  = tid & 31;
    int kstep = (tid >> 5) & 31;
    int ntile = (tid >> 10) & 3;
    int mat   = tid >> 12;
    const float* W = (mat == 0) ? Wk : ((mat == 1) ? Wq : Wv);
    int hl = lane >> 4, nl = lane & 15;
    v16h frag;
#pragma unroll
    for (int vg = 0; vg < 8; vg++) {
        int kl = kbase(hl, vg);
        int k0 = kstep * 32 + kl;
        frag[2 * vg]     = (_Float16)W[(size_t)k0 * Hn + ntile * 16 + nl];
        frag[2 * vg + 1] = (_Float16)W[(size_t)(k0 + 1) * Hn + ntile * 16 + nl];
    }
    *(v16h*)(wpack + (size_t)tid * 16) = frag;
}

// ---------------------------------------------------------------------------
// Kernel 1: fused QKV projection. 1 block = one 16-row M-tile of x (over B*T),
// 4 waves = 4 N-tiles of H=64. K = C = 1024 -> 32 WMMA k-steps per matrix.
// Outputs: qh [b*T][H] f16 (pre-scaled by C^-0.5), kh [b*T][H] f16,
//          vhT [b][H][T] f16 (transposed for contiguous PV B-fragment loads).
// ---------------------------------------------------------------------------
__global__ __launch_bounds__(128) void qkv_kernel(
    const float* __restrict__ x, const _Float16* __restrict__ wpack,
    const float* __restrict__ bk, const float* __restrict__ bq, const float* __restrict__ bv,
    _Float16* __restrict__ qh, _Float16* __restrict__ kh, _Float16* __restrict__ vhT) {
    int mtile = blockIdx.x;                 // 0 .. B*T/16 - 1
    int wave  = threadIdx.x >> 5;           // ntile 0..3
    int lane  = threadIdx.x & 31;
    int hl = lane >> 4, nl = lane & 15;
    int row0 = mtile * 16;                  // row into flattened [B*T]

    v8f ck = {}, cq = {}, cv = {};
    const _Float16* wbase = wpack + ((size_t)wave * 32) * 512 + (size_t)lane * 16;
    const size_t matStride = (size_t)4 * 32 * 32 * 16;   // 65536 halves per matrix

    for (int ks = 0; ks < 32; ks++) {
        // A fragment: 16x32 tile of x, row = row0 + nl, K = ks*32 + striping
        v16h a;
#pragma unroll
        for (int vg = 0; vg < 8; vg++) {
            int kl = kbase(hl, vg);
            f2 xv = *(const f2*)(x + (size_t)(row0 + nl) * Cn + ks * 32 + kl);
            a[2 * vg]     = (_Float16)xv.x;
            a[2 * vg + 1] = (_Float16)xv.y;
        }
        const _Float16* wp = wbase + (size_t)ks * 512;
        v16h bkf = *(const v16h*)(wp);
        v16h bqf = *(const v16h*)(wp + matStride);
        v16h bvf = *(const v16h*)(wp + 2 * matStride);
        ck = wmma_f16(a, bkf, ck);
        cq = wmma_f16(a, bqf, cq);
        cv = wmma_f16(a, bvf, cv);
    }

    float bkv = bk[wave * 16 + nl];
    float bqv = bq[wave * 16 + nl];
    float bvv = bv[wave * 16 + nl];
    int b = row0 / Tn;                       // all 16 rows share one batch (T % 16 == 0)
#pragma unroll
    for (int r = 0; r < 8; r++) {
        int t = row0 + r + 8 * hl;           // D layout: M = r + 8*hl, N = nl
        kh[(size_t)t * Hn + wave * 16 + nl] = (_Float16)(ck[r] + bkv);
        qh[(size_t)t * Hn + wave * 16 + nl] = (_Float16)((cq[r] + bqv) * 0.03125f); // C^-0.5
        int tt = t - b * Tn;
        vhT[((size_t)b * Hn + wave * 16 + nl) * Tn + tt] = (_Float16)(cv[r] + bvv);
    }
}

// ---------------------------------------------------------------------------
// Kernel 2: flash attention. 1 wave per 16-query tile; causal key chunks of 32.
// Per chunk: 4 QK WMMAs -> online softmax (16-lane shfl_xor row reductions) ->
// P via LDS re-layout (s_wait_dscnt orders cross-lane DS RAW) -> 4 PV WMMAs.
// ---------------------------------------------------------------------------
__global__ __launch_bounds__(128) void attn_kernel(
    const _Float16* __restrict__ qh, const _Float16* __restrict__ kh,
    const _Float16* __restrict__ vhT, float* __restrict__ out) {
    __shared__ __align__(16) _Float16 plds[4][16][32];

    int wave = threadIdx.x >> 5;
    int lane = threadIdx.x & 31;
    int tile = blockIdx.x * 4 + wave;        // 0 .. B*T/16 - 1
    int b  = tile / (Tn / 16);
    int q0 = (tile % (Tn / 16)) * 16;
    int hl = lane >> 4, nl = lane & 15;

    const _Float16* qb = qh  + (size_t)b * Tn * Hn;
    const _Float16* kb = kh  + (size_t)b * Tn * Hn;
    const _Float16* vb = vhT + (size_t)b * Hn * Tn;

    // Preload Q tile: two A fragments (h-dim k-steps 0..31, 32..63)
    v16h aq0, aq1;
#pragma unroll
    for (int vg = 0; vg < 8; vg++) {
        int kl = kbase(hl, vg);
        const _Float16* qr = qb + (size_t)(q0 + nl) * Hn;
        h2 t0 = *(const h2*)(qr + kl);
        h2 t1 = *(const h2*)(qr + 32 + kl);
        aq0[2 * vg] = t0.x; aq0[2 * vg + 1] = t0.y;
        aq1[2 * vg] = t1.x; aq1[2 * vg + 1] = t1.y;
    }

    v8f o0 = {}, o1 = {}, o2 = {}, o3 = {};
    float mrow[8], lrow[8];
#pragma unroll
    for (int r = 0; r < 8; r++) { mrow[r] = -__builtin_inff(); lrow[r] = 0.0f; }

    for (int c0 = 0; c0 <= q0; c0 += 32) {
        // ----- S = Q K^T for 32 keys (two 16-wide N tiles) -----
        v8f s[2];
#pragma unroll
        for (int nt = 0; nt < 2; nt++) {
            int key = c0 + nt * 16 + nl;
            const _Float16* kr = kb + (size_t)key * Hn;
            v16h bk0, bk1;
#pragma unroll
            for (int vg = 0; vg < 8; vg++) {
                int kl = kbase(hl, vg);
                h2 t0 = *(const h2*)(kr + kl);
                h2 t1 = *(const h2*)(kr + 32 + kl);
                bk0[2 * vg] = t0.x; bk0[2 * vg + 1] = t0.y;
                bk1[2 * vg] = t1.x; bk1[2 * vg + 1] = t1.y;
            }
            v8f acc = {};
            acc = wmma_f16(aq0, bk0, acc);
            acc = wmma_f16(aq1, bk1, acc);
            s[nt] = acc;
        }

        // ----- causal mask (only the diagonal chunk can mask) -----
        if (c0 + 31 > q0) {
#pragma unroll
            for (int nt = 0; nt < 2; nt++) {
                int key = c0 + nt * 16 + nl;
#pragma unroll
                for (int r = 0; r < 8; r++) {
                    int qrow = q0 + r + 8 * hl;
                    if (key > qrow) s[nt][r] = -__builtin_inff();
                }
            }
        }

        // ----- online softmax, per D-layout row (M = r + 8*hl) -----
#pragma unroll
        for (int r = 0; r < 8; r++) {
            float t = fmaxf(s[0][r], s[1][r]);
            t = fmaxf(t, __shfl_xor(t, 1, 32));
            t = fmaxf(t, __shfl_xor(t, 2, 32));
            t = fmaxf(t, __shfl_xor(t, 4, 32));
            t = fmaxf(t, __shfl_xor(t, 8, 32));
            float mnew  = fmaxf(mrow[r], t);
            float alpha = __expf(mrow[r] - mnew);
            float p0 = __expf(s[0][r] - mnew);
            float p1 = __expf(s[1][r] - mnew);
            float rs = p0 + p1;
            rs += __shfl_xor(rs, 1, 32);
            rs += __shfl_xor(rs, 2, 32);
            rs += __shfl_xor(rs, 4, 32);
            rs += __shfl_xor(rs, 8, 32);
            lrow[r] = lrow[r] * alpha + rs;
            mrow[r] = mnew;
            o0[r] *= alpha; o1[r] *= alpha; o2[r] *= alpha; o3[r] *= alpha;
            plds[wave][r + 8 * hl][nl]      = (_Float16)p0;
            plds[wave][r + 8 * hl][16 + nl] = (_Float16)p1;
        }

        // same-wave cross-lane LDS RAW: DS ops are in-order per wave; the
        // split dscnt wait + memory clobber pins the store->load ordering.
        asm volatile("s_wait_dscnt 0" ::: "memory");

        // ----- reload P as A fragment (16 x 32 keys) -----
        v16h pa;
#pragma unroll
        for (int vg = 0; vg < 8; vg++) {
            int kl = kbase(hl, vg);
            h2 t = *(const h2*)(&plds[wave][nl][kl]);
            pa[2 * vg] = t.x; pa[2 * vg + 1] = t.y;
        }

        // ----- O += P V : 4 N-tiles of H, V^T gives contiguous fragment loads
#pragma unroll
        for (int j = 0; j < 4; j++) {
            const _Float16* vr = vb + (size_t)(j * 16 + nl) * Tn + c0;
            v16h vf;
#pragma unroll
            for (int vg = 0; vg < 8; vg++) {
                int kl = kbase(hl, vg);
                h2 t = *(const h2*)(vr + kl);
                vf[2 * vg] = t.x; vf[2 * vg + 1] = t.y;
            }
            if (j == 0) o0 = wmma_f16(pa, vf, o0);
            if (j == 1) o1 = wmma_f16(pa, vf, o1);
            if (j == 2) o2 = wmma_f16(pa, vf, o2);
            if (j == 3) o3 = wmma_f16(pa, vf, o3);
        }
    }

    // ----- epilogue: divide by row sums, store f32 [B,T,H] -----
#pragma unroll
    for (int r = 0; r < 8; r++) {
        float inv = 1.0f / lrow[r];
        size_t base = ((size_t)b * Tn + q0 + r + 8 * hl) * Hn;
        out[base + 0  + nl] = o0[r] * inv;
        out[base + 16 + nl] = o1[r] * inv;
        out[base + 32 + nl] = o2[r] * inv;
        out[base + 48 + nl] = o3[r] * inv;
    }
}

// ---------------------------------------------------------------------------
extern "C" void kernel_launch(void* const* d_in, const int* in_sizes, int n_in,
                              void* d_out, int out_size, void* d_ws, size_t ws_size,
                              hipStream_t stream) {
    const float* x  = (const float*)d_in[0];
    const float* Wk = (const float*)d_in[1];
    const float* bk = (const float*)d_in[2];
    const float* Wq = (const float*)d_in[3];
    const float* bq = (const float*)d_in[4];
    const float* Wv = (const float*)d_in[5];
    const float* bv = (const float*)d_in[6];
    float* out = (float*)d_out;

    char* ws = (char*)d_ws;
    // workspace layout (bytes):
    //   wpack : 3*4*32*32*16 halves = 393216
    //   qh    : B*T*H halves       = 2097152
    //   kh    : B*T*H halves       = 2097152
    //   vhT   : B*H*T halves       = 2097152
    _Float16* wpack = (_Float16*)(ws);
    _Float16* qh   = (_Float16*)(ws + 393216);
    _Float16* kh   = (_Float16*)(ws + 393216 + 2097152);
    _Float16* vhT  = (_Float16*)(ws + 393216 + 2 * 2097152);

    repack_w<<<48, 256, 0, stream>>>(Wk, Wq, Wv, wpack);
    qkv_kernel<<<(Bn * Tn) / 16, 128, 0, stream>>>(x, wpack, bk, bq, bv, qh, kh, vhT);
    attn_kernel<<<(Bn * Tn) / (16 * 4), 128, 0, stream>>>(qh, kh, vhT, out);
}
